// CSPLayer_27693949125352
// MI455X (gfx1250) — matrix operand order repaired and checked by reading, not compiled, not run on hardware
//
#include <hip/hip_runtime.h>
#include <hip/hip_bf16.h>

// ---------------------------------------------------------------------------
// CDNA5 (gfx1250) fused CSP layer:
//   edge MLP (bf16 WMMA, f32 accum) + scatter-mean (f32 atomics, L2-resident)
//   node MLP (bf16 WMMA) + residual.
// ---------------------------------------------------------------------------

typedef __attribute__((ext_vector_type(16))) __bf16 v16bf;
typedef __attribute__((ext_vector_type(8)))  float  v8f;

#define HDIM 128

// float -> bf16 (round-to-nearest-even), stored as raw u16
__device__ __forceinline__ unsigned short f2bf(float x) {
    union { float f; unsigned int u; } v; v.f = x;
    unsigned int u = v.u;
    unsigned int r = (u + 0x7FFFu + ((u >> 16) & 1u)) >> 16;
    return (unsigned short)r;
}

__device__ __forceinline__ float silu(float x) {
    return x / (1.0f + __expf(-x));
}

// A-matrix fragment (16x32 bf16) from an LDS row-major tile.
// ISA layout: lane<16 holds row m=lane, K = {half*8..+7, 16+half*8..+7}
//  -> exactly two contiguous 16B chunks per lane.
__device__ __forceinline__ v16bf load_frag_a(const unsigned short* lds, int ldk, int kc) {
    int lane = threadIdx.x & 31;
    int half = lane >> 4;
    int m    = lane & 15;
    const unsigned short* p = lds + m * ldk + kc * 32 + half * 8;
    union { v16bf v; uint4 q[2]; } u;
    u.q[0] = *(const uint4*)(p);
    u.q[1] = *(const uint4*)(p + 16);
    return u.v;
}

// B-matrix fragment (32x16 bf16) from fragment-major packed weights:
// frag = kc*(N/16)+nt, then [lane][elem] contiguous -> 32B coalesced per lane.
__device__ __forceinline__ v16bf load_frag_b(const unsigned short* Wp, int frag) {
    const uint4* p = (const uint4*)(Wp + (size_t)frag * 512 + (threadIdx.x & 31) * 16);
    union { v16bf v; uint4 q[2]; } u;
    u.q[0] = p[0];
    u.q[1] = p[1];
    return u.v;
}

// ---------------------------------------------------------------------------
// Prep kernels
// ---------------------------------------------------------------------------

// Pack f32 weight [K x N] row-major into bf16 WMMA-B fragment layout,
// zero-padding K up to Kp (multiple of 32). One thread per packed element.
__global__ void pack_weight_kernel(const float* __restrict__ W,
                                   unsigned short* __restrict__ Wp,
                                   int K, int N, int Kp) {
    int idx   = blockIdx.x * 256 + threadIdx.x;
    int total = (Kp / 32) * (N / 16) * 512;
    if (idx >= total) return;
    int e    = idx & 15;
    int lane = (idx >> 4) & 31;
    int f    = idx >> 9;
    int ntiles = N / 16;
    int kc = f / ntiles, nt = f % ntiles;
    int half = lane >> 4, n = lane & 15;
    int row = kc * 32 + (e < 8 ? half * 8 + e : 16 + half * 8 + (e - 8));   // ISA B layout: k = {half*8..+7, 16+half*8..+7}
    int col = nt * 16 + n;
    float v = (row < K) ? W[(size_t)row * N + col] : 0.0f;
    Wp[idx] = f2bf(v);
}

// lat_ips[g, i, k] = sum_j L[g,i,j] * L[g,k,j]
__global__ void lattice_kernel(const float* __restrict__ lattices,
                               float* __restrict__ lat_ips, int G) {
    int idx = blockIdx.x * 256 + threadIdx.x;
    if (idx >= G * 9) return;
    int g = idx / 9, ij = idx % 9;
    int i = ij / 3, k = ij % 3;
    const float* L = lattices + (size_t)g * 9;
    lat_ips[idx] = L[i*3+0]*L[k*3+0] + L[i*3+1]*L[k*3+1] + L[i*3+2]*L[k*3+2];
}

__global__ void zero_kernel(float* __restrict__ p, size_t n) {
    size_t idx = (size_t)blockIdx.x * 256 + threadIdx.x;
    if (idx < n) p[idx] = 0.0f;
}

// ---------------------------------------------------------------------------
// Edge kernel: per block, 16 edges x 128 features, 8 waves (one 16x16 N-tile
// each). GEMM1 (K=288 padded) -> SiLU -> GEMM2 (K=128) -> SiLU -> store +
// scatter-add into sums/cnt.
// ---------------------------------------------------------------------------
#define LDK1 296   // 288 + 8 pad (keeps rows 16B-aligned, breaks bank stride)
#define LDK2 136   // 128 + 8 pad

__global__ __launch_bounds__(256) void edge_kernel(
    const float* __restrict__ node_features,
    const float* __restrict__ lat_ips,
    const float* __restrict__ frac_diff,
    const unsigned short* __restrict__ W1p, const float* __restrict__ b1,
    const unsigned short* __restrict__ W2p, const float* __restrict__ b2,
    const int* __restrict__ src, const int* __restrict__ dst,
    const int* __restrict__ e2g,
    float* __restrict__ edge_out, float* __restrict__ sums,
    float* __restrict__ cnt, int E)
{
    __shared__ unsigned short A1[16 * LDK1];
    __shared__ unsigned short A2[16 * LDK2];
    __shared__ float          EO[16 * HDIM];
    __shared__ int            sDst[16];

    const int e0  = blockIdx.x * 16;
    const int tid = threadIdx.x;

    // Gather hi / hj (L2-resident node_features table), convert to bf16.
    for (int idx = tid; idx < 16 * HDIM; idx += 256) {
        int e = idx >> 7, c = idx & 127;
        int ee = min(e0 + e, E - 1);
        A1[e * LDK1 + c]        = f2bf(node_features[(size_t)src[ee] * HDIM + c]);
        A1[e * LDK1 + HDIM + c] = f2bf(node_features[(size_t)dst[ee] * HDIM + c]);
    }
    // Tail columns 256..295: 9 lattice ips, 3 frac_diff, 28 zero pad.
    for (int idx = tid; idx < 16 * 40; idx += 256) {
        int e = idx / 40, c = idx % 40;
        int ee = min(e0 + e, E - 1);
        float v = 0.0f;
        if (c < 9)       v = lat_ips[(size_t)e2g[ee] * 9 + c];
        else if (c < 12) v = frac_diff[(size_t)ee * 3 + (c - 9)];
        A1[e * LDK1 + 256 + c] = f2bf(v);
    }
    if (tid < 16) sDst[tid] = dst[min(e0 + tid, E - 1)];
    __syncthreads();

    const int wave = tid >> 5;
    const int lane = tid & 31;
    const int half = lane >> 4, nl = lane & 15;
    const int col  = wave * 16 + nl;

    // GEMM1: [16,288] x [288,128], this wave's 16-col tile. 9 WMMA ops.
    v8f acc = {};
#pragma unroll
    for (int kc = 0; kc < 9; ++kc) {
        v16bf a = load_frag_a(A1, LDK1, kc);
        v16bf b = load_frag_b(W1p, kc * 8 + wave);
        acc = __builtin_amdgcn_wmma_f32_16x16x32_bf16(false, a, false, b,
                                                      (short)0, acc, false, false);
    }
    {
        float bias = b1[col];
#pragma unroll
        for (int r = 0; r < 8; ++r) {
            int m = half * 8 + r;                     // C/D layout: row = r + 8*half
            A2[m * LDK2 + col] = f2bf(silu(acc[r] + bias));
        }
    }
    __syncthreads();

    // GEMM2: [16,128] x [128,128]. 4 WMMA ops.
    v8f acc2 = {};
#pragma unroll
    for (int kc = 0; kc < 4; ++kc) {
        v16bf a = load_frag_a(A2, LDK2, kc);
        v16bf b = load_frag_b(W2p, kc * 8 + wave);
        acc2 = __builtin_amdgcn_wmma_f32_16x16x32_bf16(false, a, false, b,
                                                       (short)0, acc2, false, false);
    }
    {
        float bias = b2[col];
#pragma unroll
        for (int r = 0; r < 8; ++r) {
            int m = half * 8 + r;
            EO[m * HDIM + col] = silu(acc2[r] + bias);
        }
    }
    __syncthreads();

    // Coalesced store of edge_features + scatter-add (atomics resolve in L2:
    // sums is 25.6 MB < 192 MB global L2).
    const int valid = min(E - e0, 16);
    for (int idx = tid; idx < valid * HDIM; idx += 256) {
        int e = idx >> 7, c = idx & 127;
        float v = EO[e * HDIM + c];
        edge_out[(size_t)(e0 + e) * HDIM + c] = v;
        atomicAdd(&sums[(size_t)sDst[e] * HDIM + c], v);
    }
    if (tid < valid) atomicAdd(&cnt[sDst[tid]], 1.0f);
}

// ---------------------------------------------------------------------------
// Node kernel: 16 nodes x 128 features per block. Build [nf | mean-agg],
// GEMM3 (K=256) -> SiLU -> GEMM4 (K=128) -> SiLU -> residual add.
// ---------------------------------------------------------------------------
#define LDK3 264   // 256 + 8 pad

__global__ __launch_bounds__(256) void node_kernel(
    const float* __restrict__ node_features,
    const float* __restrict__ sums, const float* __restrict__ cnt,
    const unsigned short* __restrict__ W3p, const float* __restrict__ b3,
    const unsigned short* __restrict__ W4p, const float* __restrict__ b4,
    float* __restrict__ node_out, int N)
{
    __shared__ unsigned short A1[16 * LDK3];
    __shared__ unsigned short A2[16 * LDK2];
    __shared__ float          OO[16 * HDIM];

    const int n0  = blockIdx.x * 16;
    const int tid = threadIdx.x;

    for (int idx = tid; idx < 16 * HDIM; idx += 256) {
        int r = idx >> 7, c = idx & 127;
        int n = min(n0 + r, N - 1);
        float nf  = node_features[(size_t)n * HDIM + c];
        float agg = sums[(size_t)n * HDIM + c] / fmaxf(cnt[n], 1.0f);
        A1[r * LDK3 + c]        = f2bf(nf);
        A1[r * LDK3 + HDIM + c] = f2bf(agg);
    }
    __syncthreads();

    const int wave = tid >> 5;
    const int lane = tid & 31;
    const int half = lane >> 4, nl = lane & 15;
    const int col  = wave * 16 + nl;

    // GEMM3: [16,256] x [256,128]. 8 WMMA ops.
    v8f acc = {};
#pragma unroll
    for (int kc = 0; kc < 8; ++kc) {
        v16bf a = load_frag_a(A1, LDK3, kc);
        v16bf b = load_frag_b(W3p, kc * 8 + wave);
        acc = __builtin_amdgcn_wmma_f32_16x16x32_bf16(false, a, false, b,
                                                      (short)0, acc, false, false);
    }
    {
        float bias = b3[col];
#pragma unroll
        for (int r = 0; r < 8; ++r) {
            int m = half * 8 + r;
            A2[m * LDK2 + col] = f2bf(silu(acc[r] + bias));
        }
    }
    __syncthreads();

    // GEMM4: [16,128] x [128,128]. 4 WMMA ops.
    v8f acc2 = {};
#pragma unroll
    for (int kc = 0; kc < 4; ++kc) {
        v16bf a = load_frag_a(A2, LDK2, kc);
        v16bf b = load_frag_b(W4p, kc * 8 + wave);
        acc2 = __builtin_amdgcn_wmma_f32_16x16x32_bf16(false, a, false, b,
                                                       (short)0, acc2, false, false);
    }
    {
        float bias = b4[col];
#pragma unroll
        for (int r = 0; r < 8; ++r) {
            int m = half * 8 + r;
            int n = min(n0 + m, N - 1);
            float res = node_features[(size_t)n * HDIM + col];
            OO[m * HDIM + col] = res + silu(acc2[r] + bias);
        }
    }
    __syncthreads();

    const int valid = min(N - n0, 16);
    for (int idx = tid; idx < valid * HDIM; idx += 256) {
        int r = idx >> 7, c = idx & 127;
        node_out[(size_t)(n0 + r) * HDIM + c] = OO[r * HDIM + c];
    }
}

// ---------------------------------------------------------------------------
// Launch
// ---------------------------------------------------------------------------
extern "C" void kernel_launch(void* const* d_in, const int* in_sizes, int n_in,
                              void* d_out, int out_size, void* d_ws, size_t ws_size,
                              hipStream_t stream) {
    (void)n_in; (void)out_size; (void)ws_size;

    const float* node_features = (const float*)d_in[0];
    // d_in[1] = frac_coords (unused by reference)
    const float* lattices  = (const float*)d_in[2];
    const float* frac_diff = (const float*)d_in[3];
    const float* W1 = (const float*)d_in[4];
    const float* b1 = (const float*)d_in[5];
    const float* W2 = (const float*)d_in[6];
    const float* b2 = (const float*)d_in[7];
    const float* W3 = (const float*)d_in[8];
    const float* b3 = (const float*)d_in[9];
    const float* W4 = (const float*)d_in[10];
    const float* b4 = (const float*)d_in[11];
    const int* edge_index = (const int*)d_in[12];
    const int* edge2graph = (const int*)d_in[13];

    const int N = in_sizes[0] / HDIM;
    const int G = in_sizes[2] / 9;
    const int E = in_sizes[13];

    const int* src = edge_index;
    const int* dst = edge_index + E;

    // ---- workspace layout (256B aligned) ----
    char* ws = (char*)d_ws;
    size_t off = 0;
    auto alloc = [&](size_t bytes) { char* p = ws + off; off = (off + bytes + 255) & ~(size_t)255; return p; };
    float*          sums    = (float*)alloc((size_t)N * HDIM * sizeof(float));
    float*          cnt     = (float*)alloc((size_t)N * sizeof(float));
    float*          lat_ips = (float*)alloc((size_t)G * 9 * sizeof(float));
    unsigned short* W1p     = (unsigned short*)alloc((size_t)9 * 8 * 512 * 2); // K=288
    unsigned short* W2p     = (unsigned short*)alloc((size_t)4 * 8 * 512 * 2); // K=128
    unsigned short* W3p     = (unsigned short*)alloc((size_t)8 * 8 * 512 * 2); // K=256
    unsigned short* W4p     = (unsigned short*)alloc((size_t)4 * 8 * 512 * 2); // K=128

    float* node_out = (float*)d_out;
    float* edge_out = (float*)d_out + (size_t)N * HDIM;

    // ---- prep: zero scatter buffers, pack weights, lattice inner products ----
    {
        size_t nz = (size_t)N * HDIM;
        zero_kernel<<<(unsigned)((nz + 255) / 256), 256, 0, stream>>>(sums, nz);
        zero_kernel<<<(unsigned)((N + 255) / 256), 256, 0, stream>>>(cnt, (size_t)N);
    }
    pack_weight_kernel<<<(9 * 8 * 512 + 255) / 256, 256, 0, stream>>>(W1, W1p, 268, HDIM, 288);
    pack_weight_kernel<<<(4 * 8 * 512 + 255) / 256, 256, 0, stream>>>(W2, W2p, 128, HDIM, 128);
    pack_weight_kernel<<<(8 * 8 * 512 + 255) / 256, 256, 0, stream>>>(W3, W3p, 256, HDIM, 256);
    pack_weight_kernel<<<(4 * 8 * 512 + 255) / 256, 256, 0, stream>>>(W4, W4p, 128, HDIM, 128);
    lattice_kernel<<<(G * 9 + 255) / 256, 256, 0, stream>>>(lattices, lat_ips, G);

    // ---- edge model + scatter ----
    edge_kernel<<<(E + 15) / 16, 256, 0, stream>>>(
        node_features, lat_ips, frac_diff, W1p, b1, W2p, b2,
        src, dst, edge2graph, edge_out, sums, cnt, E);

    // ---- node model ----
    node_kernel<<<(N + 15) / 16, 256, 0, stream>>>(
        node_features, sums, cnt, W3p, b3, W4p, b4, node_out, N);
}
